// LSTM_47510928229135
// MI455X (gfx1250) — matrix operand order repaired and checked by reading, not compiled, not run on hardware
//
#include <hip/hip_runtime.h>
#include <hip/hip_bf16.h>

// ---------------------------------------------------------------------------
// LSTM forward on MI455X (gfx1250, wave32, WMMA + async-LDS copies).
//   B=32, T=2048, I=256, H=512, 4H=2048.
//   1) init_state: zero bf16 h ping-pong buffers + grid-barrier counters.
//   2) wx_gemm:   wx[b*T+t, 4H] = x @ W^T + b_w   (bf16 WMMA, f32 accum)
//   3) lstm_scan: persistent kernel, 32 WGs, 1 grid barrier per timestep.
//      h ping-pong lives in global memory as bf16 and is staged into LDS
//      each step with GLOBAL_LOAD_ASYNC_TO_LDS_B128 (ASYNCcnt-tracked DMA).
// Workspace layout:
//   [0, 65536*2048) f32            wx
//   then 2*32*512 bf16             h ping-pong
//   then 4 x u32                   barrier (cnt, gen)
// ---------------------------------------------------------------------------

typedef __attribute__((ext_vector_type(16))) __bf16 v16bf;
typedef __attribute__((ext_vector_type(8)))  float  v8f;

union BF16x16 { v16bf v; unsigned u[8]; };
union BFPack2 { __bf16 h[2]; unsigned u; };

#define LSTM_B   32
#define LSTM_T   2048
#define LSTM_I   256
#define LSTM_H   512
#define LSTM_G4  2048        // 4*H
#define NWG      32          // workgroups in the persistent scan kernel

// -------------------------------- init ------------------------------------
__global__ void init_state(unsigned* hbuf_u32, unsigned* bar) {
  int i = blockIdx.x * blockDim.x + threadIdx.x;
  if (i < (2 * LSTM_B * LSTM_H) / 2) hbuf_u32[i] = 0u;   // bf16 zeros
  if (i < 4) bar[i] = 0u;
}

// ----------------------------- wx = x @ W^T + b_w -------------------------
// grid: (65536/32, 2048/256), block 256 (8 waves).
// wave w: m-tile = w&1 (16 rows), n-group = w>>1 (64 cols = 4 N-tiles).
__global__ __launch_bounds__(256) void wx_gemm(
    const float* __restrict__ x, const float* __restrict__ W,
    const float* __restrict__ bw, float* __restrict__ wx)
{
  const int lane = threadIdx.x & 31;
  const int wave = threadIdx.x >> 5;
  const int mt = wave & 1, ng = wave >> 1;
  const int m0 = blockIdx.x * 32 + mt * 16;
  const int n0 = blockIdx.y * 256 + ng * 64;
  const int half = lane >> 4;
  const int mrow = m0 + (lane & 15);
  const float* xrow = x + (size_t)mrow * LSTM_I;

  v8f acc[4] = { {}, {}, {}, {} };

  for (int k0 = 0; k0 < LSTM_I; k0 += 32) {
    __builtin_prefetch(xrow + k0 + 32, 0, 1);
    // A fragment: 16 contiguous K per lane, lane half adds +16 to K (matches B)
    BF16x16 a;
#pragma unroll
    for (int j = 0; j < 8; ++j) {
      int kp = 2 * j + half * 16;
      float2 f = *(const float2*)(xrow + k0 + kp);
      a.v[2 * j]     = (__bf16)f.x;      // native v_cvt f32->bf16
      a.v[2 * j + 1] = (__bf16)f.y;
    }
#pragma unroll
    for (int nt = 0; nt < 4; ++nt) {
      // B fragment: lane = column, 16 contiguous K, lane half adds +16 to K
      const int ncol = n0 + nt * 16 + (lane & 15);
      const float* wrow = W + (size_t)ncol * LSTM_I + k0 + half * 16;
      BF16x16 bm;
#pragma unroll
      for (int q = 0; q < 4; ++q) {
        float4 f = *(const float4*)(wrow + q * 4);
        bm.v[4 * q]     = (__bf16)f.x;
        bm.v[4 * q + 1] = (__bf16)f.y;
        bm.v[4 * q + 2] = (__bf16)f.z;
        bm.v[4 * q + 3] = (__bf16)f.w;
      }
      acc[nt] = __builtin_amdgcn_wmma_f32_16x16x32_bf16(
          false, a.v, false, bm.v, (short)0, acc[nt], false, false);
    }
  }
  // D tile layout: lane%16 = col, rows r + (lane/16)*8; add bias, store f32.
#pragma unroll
  for (int nt = 0; nt < 4; ++nt) {
    int n = n0 + nt * 16 + (lane & 15);
    float bb = bw[n];
#pragma unroll
    for (int r = 0; r < 8; ++r) {
      int m = m0 + r + half * 8;
      wx[(size_t)m * LSTM_G4 + n] = acc[nt][r] + bb;
    }
  }
}

// ------------------------------ recurrent scan ----------------------------
// 32 WGs x 256 threads, persistent. Dynamic LDS:
//   Ul:  bf16 [64][520]   offset 0       (66560 B)  U slice, resident
//   hl:  bf16 [32][520]   offset 66560   (33280 B)  h_t staged per step (async DMA)
//   gl:  f32  [4][32][16] offset 99840   ( 8192 B)  per-step gate partials
//   bul: f32  [64]        offset 108032  (  256 B)  b_u slice
#define LDS_BYTES 108288

__global__ __launch_bounds__(256) void lstm_scan(
    const float* __restrict__ wx, const float* __restrict__ U,
    const float* __restrict__ bu, float* __restrict__ out,
    __bf16* __restrict__ hbuf, unsigned* __restrict__ bar)
{
  extern __shared__ char smem[];
  __bf16* Ul  = (__bf16*)smem;                // [64][520]
  __bf16* hl  = (__bf16*)(smem + 66560);      // [32][520]
  float*  gl  = (float*)(smem + 99840);       // [4][32][16]
  float*  bul = (float*)(smem + 108032);      // [64]

  const int tid  = threadIdx.x;
  const int lane = tid & 31;
  const int wave = tid >> 5;
  const int wg   = blockIdx.x;          // 0..31, owns hidden units [wg*16, wg*16+16)

  // Stage U slice once: row r -> gate (r/16), unit (r%16); bf16, padded stride 520.
  for (int idx = tid; idx < 64 * LSTM_H; idx += 256) {
    int r = idx >> 9, k = idx & (LSTM_H - 1);
    int gcol = (r >> 4) * LSTM_H + wg * 16 + (r & 15);
    Ul[r * 520 + k] = (__bf16)U[(size_t)gcol * LSTM_H + k];
  }
  if (tid < 64) {
    int gcol = (tid >> 4) * LSTM_H + wg * 16 + (tid & 15);
    bul[tid] = bu[gcol];
  }

  // Elementwise ownership: b = tid&31, units u_ew, u_ew+1; c in registers.
  const int b_ew = tid & 31;
  const int u_ew = (tid >> 5) * 2;
  float c0 = 0.0f, c1 = 0.0f;

  // WMMA tile ownership: 8 waves = 2 M-tiles x 4 gates.
  const int mt = wave & 1;
  const int gt = wave >> 1;

  volatile unsigned* cnt = bar;
  volatile unsigned* gen = bar + 1;

  // 32-bit LDS byte address of hl (flat shared pointers carry the LDS offset
  // in their low 32 bits per the gfx1250 aperture mapping).
  const unsigned hl_base = (unsigned)(uintptr_t)hl;

  __syncthreads();

  for (int t = 0; t < LSTM_T; ++t) {
    // ---- stage h_t: bf16 global -> bf16 LDS via async DMA (ASYNCcnt) ----
    const __bf16* hsrc = hbuf + (size_t)(t & 1) * (LSTM_B * LSTM_H);
    for (int c = tid; c < (LSTM_B * LSTM_H) / 8; c += 256) {   // 16B chunks
      int b  = c >> 6;            // 64 chunks per batch row
      int k8 = (c & 63) * 8;
      unsigned lds_addr = hl_base + (unsigned)(b * 520 + k8) * 2u;
      unsigned goff     = (unsigned)(b * LSTM_H + k8) * 2u;
      asm volatile("global_load_async_to_lds_b128 %0, %1, %2"
                   :: "v"(lds_addr), "v"(goff), "s"(hsrc)
                   : "memory");
    }
    asm volatile("s_wait_asynccnt 0" ::: "memory");
    __syncthreads();

    // ---- gates_part = h @ U_slice^T via WMMA ----
    v8f acc = {};
#pragma unroll 4
    for (int kc = 0; kc < LSTM_H / 32; ++kc) {
      int k0 = kc * 32;
      BF16x16 a;
      {
        int m = mt * 16 + (lane & 15);
        int hofs = (lane >> 4) * 16;
        const unsigned* hrow = (const unsigned*)(hl + m * 520 + k0);
#pragma unroll
        for (int j = 0; j < 8; ++j) {
          int kp = 2 * j + hofs;
          a.u[j] = hrow[kp >> 1];
        }
      }
      BF16x16 bm;
      {
        int r = gt * 16 + (lane & 15);
        int kbase = k0 + (lane >> 4) * 16;
        const uint4* p = (const uint4*)(Ul + r * 520 + kbase);
        uint4 q0 = p[0], q1 = p[1];
        bm.u[0] = q0.x; bm.u[1] = q0.y; bm.u[2] = q0.z; bm.u[3] = q0.w;
        bm.u[4] = q1.x; bm.u[5] = q1.y; bm.u[6] = q1.z; bm.u[7] = q1.w;
      }
      acc = __builtin_amdgcn_wmma_f32_16x16x32_bf16(
          false, a.v, false, bm.v, (short)0, acc, false, false);
    }
    // spill D tile to LDS: gl[gate][batch][unit]
    {
      int n = lane & 15;
      int mbase = mt * 16 + (lane >> 4) * 8;
#pragma unroll
      for (int r = 0; r < 8; ++r)
        gl[(gt * 32 + mbase + r) * 16 + n] = acc[r];
    }
    __syncthreads();

    // ---- elementwise: gates -> c,h ; h to out (f32) + ping-pong (bf16) ----
    {
      const size_t wxrow = ((size_t)b_ew * LSTM_T + t) * LSTM_G4;
      const size_t orow  = ((size_t)b_ew * LSTM_T + t) * LSTM_H;
      __bf16* hdst = hbuf + (size_t)((t + 1) & 1) * (LSTM_B * LSTM_H);
      float hout[2];
#pragma unroll
      for (int uu = 0; uu < 2; ++uu) {
        int u = u_ew + uu;
        int col0 = wg * 16 + u;
        float ig = gl[(0 * 32 + b_ew) * 16 + u] + wx[wxrow + col0       ] + bul[ 0 + u];
        float fg = gl[(1 * 32 + b_ew) * 16 + u] + wx[wxrow + col0 + 512 ] + bul[16 + u];
        float gg = gl[(2 * 32 + b_ew) * 16 + u] + wx[wxrow + col0 + 1024] + bul[32 + u];
        float og = gl[(3 * 32 + b_ew) * 16 + u] + wx[wxrow + col0 + 1536] + bul[48 + u];
        ig = 1.0f / (1.0f + __expf(-ig));
        fg = 1.0f / (1.0f + __expf(-fg));
        og = 1.0f / (1.0f + __expf(-og));
        gg = tanhf(gg);
        float c = fg * (uu ? c1 : c0) + ig * gg;
        if (uu) c1 = c; else c0 = c;
        float h = og * tanhf(c);
        hout[uu] = h;
        out[orow + col0] = h;
      }
      BFPack2 pk;
      pk.h[0] = (__bf16)hout[0];
      pk.h[1] = (__bf16)hout[1];
      ((unsigned*)hdst)[(b_ew * LSTM_H + wg * 16 + u_ew) >> 1] = pk.u;
    }
    __syncthreads();

    // ---- device-wide barrier (phase counter in ws) ----
    if (tid == 0) {
      __threadfence();
      unsigned p = *gen;
      unsigned prev = atomicAdd((unsigned*)cnt, 1u);
      if (prev == NWG - 1) {
        *cnt = 0u;
        __threadfence();
        atomicAdd((unsigned*)gen, 1u);
      } else {
        while (*gen == p) { __builtin_amdgcn_s_sleep(2); }
      }
      __threadfence();
    }
    __syncthreads();
  }
}

// ------------------------------- launch -----------------------------------
extern "C" void kernel_launch(void* const* d_in, const int* in_sizes, int n_in,
                              void* d_out, int out_size, void* d_ws, size_t ws_size,
                              hipStream_t stream) {
  const float* x   = (const float*)d_in[0];
  const float* W   = (const float*)d_in[1];
  const float* b_w = (const float*)d_in[2];
  const float* U   = (const float*)d_in[3];
  const float* b_u = (const float*)d_in[4];
  float* out = (float*)d_out;

  const size_t wx_elems = (size_t)LSTM_B * LSTM_T * LSTM_G4;   // 512 MB f32
  float*    wx   = (float*)d_ws;
  __bf16*   hbuf = (__bf16*)(wx + wx_elems);                   // 2*B*H bf16
  unsigned* bar  = (unsigned*)(hbuf + 2 * LSTM_B * LSTM_H);

  init_state<<<128, 256, 0, stream>>>((unsigned*)hbuf, bar);

  dim3 g1((LSTM_B * LSTM_T) / 32, LSTM_G4 / 256);
  wx_gemm<<<g1, 256, 0, stream>>>(x, W, b_w, wx);

  lstm_scan<<<NWG, 256, LDS_BYTES, stream>>>(wx, U, b_u, out, hbuf, bar);
}
